// RGCNBasisLayer_45732811768294
// MI455X (gfx1250) — matrix operand order repaired and checked by reading, not compiled, not run on hardware
//
#include <hip/hip_runtime.h>
#include <hip/hip_bf16.h>
#include <math.h>

// ---------------- problem constants (from reference) ----------------
#define N_NODES   50000
#define N_EDGES   800000
#define INP_DIM   32
#define OUT_DIM   32
#define ATTN_DIM  32
#define NUM_RELS  86
#define NUM_BASES 16
#define INV_SCALE 0.17677669529663689f   // 1/sqrt(32)

#define NODE_TILES (N_NODES / 16)        // 3125, exact
#define BPR        16                     // blocks per relation in edge kernel

typedef __attribute__((ext_vector_type(16))) __bf16 v16bf;
typedef __attribute__((ext_vector_type(8)))  float  v8f;

union V16 { v16bf v; __bf16 e[16]; };
union V8F { v8f  v; float  e[8]; };

// ---------------- K0: zero accumulators ----------------
__global__ __launch_bounds__(256) void k0_init(float* __restrict__ nei,
                                               int* __restrict__ in_deg,
                                               int* __restrict__ last_e,
                                               int* __restrict__ rel_cnt) {
  int i = blockIdx.x * blockDim.x + threadIdx.x;
  if (i < N_NODES * OUT_DIM) nei[i] = 0.0f;
  if (i < N_NODES) { in_deg[i] = 0; last_e[i] = 0; }
  if (i < NUM_RELS) rel_cnt[i] = 0;
}

// ---------------- K1: compose W_r = sum_b c[r,b] * W_b  (bf16 out) ----------------
__global__ __launch_bounds__(256) void k1_compose(const float* __restrict__ weight,
                                                  const float* __restrict__ w_comp,
                                                  __bf16* __restrict__ wrel_bf) {
  int idx = blockIdx.x * blockDim.x + threadIdx.x;
  if (idx >= NUM_RELS * INP_DIM * OUT_DIM) return;
  int r = idx >> 10;          // / (32*32)
  int k = idx & 1023;         // i*32 + o
  float s = 0.0f;
  #pragma unroll
  for (int b = 0; b < NUM_BASES; ++b)
    s += w_comp[r * NUM_BASES + b] * weight[b * 1024 + k];
  wrel_bf[idx] = (__bf16)s;
}

// ---------------- K2: p = h@A1^T + b1, q = h@A2^T + b2 (WMMA) ----------------
__global__ __launch_bounds__(128) void k2_pq(const float* __restrict__ h,
                                             const float* __restrict__ A1w,
                                             const float* __restrict__ A1b,
                                             const float* __restrict__ A2w,
                                             const float* __restrict__ A2b,
                                             float* __restrict__ p,
                                             float* __restrict__ q) {
  __shared__ __bf16 ht[4][512];
  const int tid = threadIdx.x, wave = tid >> 5, lane = tid & 31;
  const int tile = blockIdx.x * 4 + wave;
  if (tile >= NODE_TILES) return;                 // wave-uniform guard
  const int base = tile * 16;
  const int M = lane & 15;
  const int kb = (lane >> 4) << 3;

  for (int i = lane; i < 512; i += 32)            // coalesced stage, bf16 convert
    ht[wave][i] = (__bf16)h[base * 32 + i];
  asm volatile("s_wait_dscnt 0" ::: "memory");
  __builtin_amdgcn_wave_barrier();

  V16 afr;
  #pragma unroll
  for (int e2 = 0; e2 < 16; ++e2) {
    int K = kb + e2 + ((e2 >= 8) ? 8 : 0);
    afr.e[e2] = ht[wave][M * 32 + K];
  }

  #pragma unroll
  for (int m = 0; m < 2; ++m) {
    const float* Aw = m ? A2w : A1w;
    const float* Ab = m ? A2b : A1b;
    float* outp = m ? q : p;
    #pragma unroll
    for (int half = 0; half < 2; ++half) {
      int n0 = half * 16;
      V16 bfr;
      #pragma unroll
      for (int e2 = 0; e2 < 16; ++e2)             // B[k=kb+e2(+8)][n0+M] = Aw[(n0+M)][k]
        bfr.e[e2] = (__bf16)Aw[(n0 + M) * 32 + kb + e2 + ((e2 >= 8) ? 8 : 0)];
      V8F acc;
      float bias = Ab[n0 + M];
      #pragma unroll
      for (int rr = 0; rr < 8; ++rr) acc.e[rr] = bias;
      acc.v = __builtin_amdgcn_wmma_f32_16x16x32_bf16(false, afr.v, false, bfr.v,
                                                      (short)0, acc.v, false, false);
      #pragma unroll
      for (int rr = 0; rr < 8; ++rr) {
        int row = rr + ((lane >> 4) << 3);
        outp[(base + row) * 32 + n0 + M] = acc.e[rr];
      }
    }
  }
}

// ---------------- K3: histogram edges by relation ----------------
__global__ __launch_bounds__(256) void k3_hist(const int* __restrict__ etype,
                                               int* __restrict__ rel_cnt) {
  __shared__ int hc[NUM_RELS];
  int tid = threadIdx.x;
  if (tid < NUM_RELS) hc[tid] = 0;
  __syncthreads();
  int i = blockIdx.x * blockDim.x + tid;
  if (i < N_EDGES) atomicAdd(&hc[etype[i]], 1);
  __syncthreads();
  if (tid < NUM_RELS) { int v = hc[tid]; if (v) atomicAdd(&rel_cnt[tid], v); }
}

// ---------------- K4: tiny exclusive scan (86 values) ----------------
__global__ __launch_bounds__(32) void k4_scan(const int* __restrict__ rel_cnt,
                                              int* __restrict__ rel_off,
                                              int* __restrict__ cursor) {
  if (threadIdx.x == 0) {
    int run = 0;
    for (int r = 0; r < NUM_RELS; ++r) {
      rel_off[r] = run; cursor[r] = run; run += rel_cnt[r];
    }
    rel_off[NUM_RELS] = run;
  }
}

// ---------------- K5: scatter edge ids into per-relation bins ----------------
__global__ __launch_bounds__(256) void k5_scatter(const int* __restrict__ etype,
                                                  int* __restrict__ cursor,
                                                  int* __restrict__ edge_ids) {
  int e = blockIdx.x * blockDim.x + threadIdx.x;
  if (e < N_EDGES) {
    int pos = atomicAdd(&cursor[etype[e]], 1);
    edge_ids[pos] = e;
  }
}

// ---------------- K6: fused edge kernel (attention + WMMA message + scatter) ----------------
__global__ __launch_bounds__(128) void k6_edge(const float* __restrict__ h,
                                               const float* __restrict__ p,
                                               const float* __restrict__ q,
                                               const float* __restrict__ rel_emb,
                                               const __bf16* __restrict__ wrel_bf,
                                               const int* __restrict__ src,
                                               const int* __restrict__ dst,
                                               const int* __restrict__ rel_off,
                                               const int* __restrict__ edge_ids,
                                               float* __restrict__ nei,
                                               int* __restrict__ in_deg,
                                               int* __restrict__ last_e) {
  __shared__ __bf16 wrel[1024];      // composed relation weight (bf16), shared by block
  __shared__ __bf16 xt[4][512];      // per-wave A tile: a_e * x_e
  __shared__ int    dstt[4][16];     // per-wave dst of each tile row
  const int r   = blockIdx.x % NUM_RELS;
  const int j   = blockIdx.x / NUM_RELS;   // 0..BPR-1
  const int tid = threadIdx.x, wave = tid >> 5, lane = tid & 31;
  const int M = lane & 15;
  const int kb = (lane >> 4) << 3;

  for (int i = tid; i < 1024; i += 128) wrel[i] = wrel_bf[r * 1024 + i];
  __syncthreads();

  // --- hoist: B fragments are block-invariant (W_r rows K=kb+e2(+8), cols n0+M) ---
  V16 bfr0, bfr1;
  #pragma unroll
  for (int e2 = 0; e2 < 16; ++e2) {
    bfr0.e[e2] = wrel[(kb + e2 + ((e2 >= 8) ? 8 : 0)) * 32 + M];
    bfr1.e[e2] = wrel[(kb + e2 + ((e2 >= 8) ? 8 : 0)) * 32 + 16 + M];
  }

  const float relv = rel_emb[r * 32 + lane];
  const int e0  = rel_off[r];
  const int cnt = rel_off[r + 1] - e0;
  const int ntiles = (cnt + 15) >> 4;

  for (int tile = j * 4 + wave; tile < ntiles; tile += BPR * 4) {
    // --- attention + gather: one edge per iteration, 32 lanes across feature dim ---
    #pragma unroll 4
    for (int t = 0; t < 16; ++t) {
      int ei = tile * 16 + t;                    // uniform across lanes
      if (ei < cnt) {
        int e = edge_ids[e0 + ei];
        int s = src[e], d = dst[e];
        float prod = p[s * 32 + lane] * (q[d * 32 + lane] + relv);
        #pragma unroll
        for (int off = 16; off > 0; off >>= 1) prod += __shfl_xor(prod, off, 32);
        // gamma=0: where(tanh(relu(z))>0) == z>0
        float a = (prod > 0.0f) ? tanhf(prod * INV_SCALE) : 0.0f;
        xt[wave][t * 32 + lane] = (__bf16)(a * h[s * 32 + lane]);
        if (lane == 0) {
          dstt[wave][t] = d;
          atomicAdd(&in_deg[d], 1);
          atomicMax(&last_e[d], e);
        }
      } else {
        xt[wave][t * 32 + lane] = (__bf16)0.0f;
        if (lane == 0) dstt[wave][t] = -1;
      }
    }
    asm volatile("s_wait_dscnt 0" ::: "memory");
    __builtin_amdgcn_wave_barrier();

    // --- A fragment (16-bit 16x32 layout, ISA 7.12.2) ---
    V16 afr;
    #pragma unroll
    for (int e2 = 0; e2 < 16; ++e2) {
      int K = kb + e2 + ((e2 >= 8) ? 8 : 0);
      afr.e[e2] = xt[wave][M * 32 + K];
    }

    // --- msg tile = (a*x) @ W_r via 2x WMMA, scatter with f32 atomics ---
    #pragma unroll
    for (int half = 0; half < 2; ++half) {
      int n0 = half * 16;
      V8F acc;
      #pragma unroll
      for (int rr = 0; rr < 8; ++rr) acc.e[rr] = 0.0f;
      acc.v = __builtin_amdgcn_wmma_f32_16x16x32_bf16(false, afr.v, false,
                                                      half ? bfr1.v : bfr0.v,
                                                      (short)0, acc.v, false, false);
      #pragma unroll
      for (int rr = 0; rr < 8; ++rr) {
        int row = rr + ((lane >> 4) << 3);
        int d = dstt[wave][row];
        if (d >= 0) atomicAdd(&nei[d * 32 + n0 + M], acc.e[rr]);
      }
    }
    __builtin_amdgcn_wave_barrier();
  }
}

// ---------------- K7: finalize: relu(mask * (nei + h[src[last_e]] @ W_self)) ----------------
__global__ __launch_bounds__(128) void k7_final(const float* __restrict__ h,
                                                const float* __restrict__ slw,
                                                const int* __restrict__ src,
                                                const int* __restrict__ last_e,
                                                const int* __restrict__ in_deg,
                                                float* __restrict__ out) {
  __shared__ __bf16 xs[4][512];
  const int tid = threadIdx.x, wave = tid >> 5, lane = tid & 31;
  const int tile = blockIdx.x * 4 + wave;
  if (tile >= NODE_TILES) return;                // wave-uniform guard
  const int base = tile * 16;
  const int M = lane & 15;
  const int kb = (lane >> 4) << 3;

  for (int i = lane; i < 512; i += 32) {
    int nl = i >> 5, col = i & 31;
    int e = last_e[base + nl];
    int s = src[e];
    xs[wave][i] = (__bf16)h[s * 32 + col];
  }
  asm volatile("s_wait_dscnt 0" ::: "memory");
  __builtin_amdgcn_wave_barrier();

  V16 afr;
  #pragma unroll
  for (int e2 = 0; e2 < 16; ++e2) {
    int K = kb + e2 + ((e2 >= 8) ? 8 : 0);
    afr.e[e2] = xs[wave][M * 32 + K];
  }

  #pragma unroll
  for (int half = 0; half < 2; ++half) {
    int n0 = half * 16;
    V16 bfr;
    #pragma unroll
    for (int e2 = 0; e2 < 16; ++e2)              // B[k=kb+e2(+8)][n0+M] = slw[k][n]
      bfr.e[e2] = (__bf16)slw[(kb + e2 + ((e2 >= 8) ? 8 : 0)) * 32 + n0 + M];
    V8F acc;
    #pragma unroll
    for (int rr = 0; rr < 8; ++rr) {             // C := accumulated neighbor sums
      int row = rr + ((lane >> 4) << 3);
      acc.e[rr] = out[(base + row) * 32 + n0 + M];
    }
    acc.v = __builtin_amdgcn_wmma_f32_16x16x32_bf16(false, afr.v, false, bfr.v,
                                                    (short)0, acc.v, false, false);
    #pragma unroll
    for (int rr = 0; rr < 8; ++rr) {
      int row = rr + ((lane >> 4) << 3);
      int node = base + row;
      float v = (in_deg[node] > 0) ? fmaxf(acc.e[rr], 0.0f) : 0.0f;
      out[node * 32 + n0 + M] = v;
    }
  }
}

// ---------------- host-side launch ----------------
extern "C" void kernel_launch(void* const* d_in, const int* in_sizes, int n_in,
                              void* d_out, int out_size, void* d_ws, size_t ws_size,
                              hipStream_t stream) {
  (void)in_sizes; (void)n_in; (void)out_size; (void)ws_size;
  const float* h       = (const float*)d_in[0];
  const float* weight  = (const float*)d_in[1];
  const float* w_comp  = (const float*)d_in[2];
  const float* slw     = (const float*)d_in[3];
  const float* A1w     = (const float*)d_in[4];
  const float* A1b     = (const float*)d_in[5];
  const float* A2w     = (const float*)d_in[6];
  const float* A2b     = (const float*)d_in[7];
  const float* rel_emb = (const float*)d_in[8];
  const int*   src     = (const int*)d_in[9];
  const int*   dst     = (const int*)d_in[10];
  const int*   etype   = (const int*)d_in[11];
  float* out = (float*)d_out;

  // workspace carve-out (~17 MB total)
  char* ws = (char*)d_ws;
  size_t off = 0;
  auto take = [&](size_t bytes) -> void* {
    void* ptr = ws + off;
    off = (off + bytes + 255) & ~(size_t)255;
    return ptr;
  };
  float*  p        = (float*)take((size_t)N_NODES * 32 * 4);
  float*  q        = (float*)take((size_t)N_NODES * 32 * 4);
  __bf16* wrel_bf  = (__bf16*)take((size_t)NUM_RELS * 1024 * 2);
  int*    rel_cnt  = (int*)take(NUM_RELS * 4);
  int*    rel_off  = (int*)take((NUM_RELS + 1) * 4);
  int*    cursor   = (int*)take(NUM_RELS * 4);
  int*    edge_ids = (int*)take((size_t)N_EDGES * 4);
  int*    in_deg   = (int*)take((size_t)N_NODES * 4);
  int*    last_e   = (int*)take((size_t)N_NODES * 4);

  k0_init<<<(N_NODES * OUT_DIM + 255) / 256, 256, 0, stream>>>(out, in_deg, last_e, rel_cnt);
  k1_compose<<<(NUM_RELS * 1024 + 255) / 256, 256, 0, stream>>>(weight, w_comp, wrel_bf);
  k2_pq<<<(NODE_TILES + 3) / 4, 128, 0, stream>>>(h, A1w, A1b, A2w, A2b, p, q);
  k3_hist<<<(N_EDGES + 255) / 256, 256, 0, stream>>>(etype, rel_cnt);
  k4_scan<<<1, 32, 0, stream>>>(rel_cnt, rel_off, cursor);
  k5_scatter<<<(N_EDGES + 255) / 256, 256, 0, stream>>>(etype, cursor, edge_ids);
  k6_edge<<<NUM_RELS * BPR, 128, 0, stream>>>(h, p, q, rel_emb, wrel_bf, src, dst,
                                              rel_off, edge_ids, out, in_deg, last_e);
  k7_final<<<(NODE_TILES + 3) / 4, 128, 0, stream>>>(h, slw, src, last_e, in_deg, out);
}
